// NaiveLSTM_3504693313923
// MI455X (gfx1250) — matrix operand order repaired and checked by reading, not compiled, not run on hardware
//
#include <hip/hip_runtime.h>
#include <hip/hip_bf16.h>
#include <stddef.h>

// ---------------------------------------------------------------------------
// NaiveLSTM on MI455X (gfx1250): persistent batch-parallel scan kernel.
//   B=64, S=2048, E=256, H=256.
//   - 4 independent workgroups, 16 batch rows each (WMMA M-tile), no grid sync
//   - [x_t | h] (16x512) @ Wcat (512x1024) via v_wmma_f32_16x16x32_bf16
//   - Weights pre-packed once into CDNA5 B-fragment lane layout (bf16),
//     streamed from L2 each step. An opaque SGPR offset (asm laundered per
//     time-step) stops the compiler hoisting 512 VGPRs of weights out of the
//     loop (scratch spill) while keeping global address-space provenance so
//     the loads lower to global_load_b128 (not flat_load: no DScnt coupling).
//   - x_{t+1} global loads issued before the GEMM of step t (latency hidden),
//     staged into a double-buffered LDS slab as coalesced b128/b64 traffic.
//   - Gate fusion in registers: each wave owns matching 16 columns of i,f,g,o;
//     biases folded into the WMMA accumulator init.
// ---------------------------------------------------------------------------

typedef __bf16 bf16_t;
typedef bf16_t v16bf __attribute__((ext_vector_type(16)));
typedef bf16_t v4bf  __attribute__((ext_vector_type(4)));
typedef float  v8f   __attribute__((ext_vector_type(8)));

#define SEQ_T   2048
#define E_DIM   256
#define H_DIM   256
#define KCAT    512                 // E + H
#define NGATE   1024                // 4*H
#define BATCH   64
#define ROWS_WG 16                  // WMMA M tile
#define NUM_WG  (BATCH / ROWS_WG)   // 4
#define THREADS 512                 // 16 waves (wave32)
#define KT_TILES (KCAT / 32)        // 16 K-tiles of 32
#define WC_ELEMS (KCAT * NGATE)     // 524288 bf16
#define GATE_NT_STRIDE (16 * 16 * 512)  // 16 n-tiles * 16 k-tiles * 512 elems

// ---------------------------------------------------------------------------
// One-time prologue: fuse the 8 weight matrices into Wcat (512x1024) bf16,
// pre-swizzled into the WMMA B-fragment layout (ISA 7.12.2, 16-bit B 32x16:
// lanes 0-15 hold K=0..7,16..23, lanes 16-31 hold K=8..15,24..31; column = lane % 16).
// Block (nt,kt) is 512 contiguous bf16: [lane][half]. Also fuses the biases.
// ---------------------------------------------------------------------------
__global__ __launch_bounds__(256)
void lstm_pack_weights(const float* __restrict__ Wii, const float* __restrict__ Whi, const float* __restrict__ bi,
                       const float* __restrict__ Wif, const float* __restrict__ Whf, const float* __restrict__ bf_,
                       const float* __restrict__ Wig, const float* __restrict__ Whg, const float* __restrict__ bg,
                       const float* __restrict__ Wio, const float* __restrict__ Who, const float* __restrict__ bo,
                       bf16_t* __restrict__ Wc, float* __restrict__ biasc)
{
    int e = blockIdx.x * blockDim.x + threadIdx.x;
    if (e < WC_ELEMS) {
        int bidx  = e >> 9;             // nt*16 + kt
        int inner = e & 511;
        int lane  = inner >> 4;
        int half  = inner & 15;
        int kt = bidx & 15;
        int nt = bidx >> 4;
        int k_in = (half & 7) + ((half >= 8) ? 16 : 0) + ((lane >= 16) ? 8 : 0);   // B-fragment K mapping
        int n_in = lane & 15;
        int r = kt * 32 + k_in;         // concat K row: [0,256)=x, [256,512)=h
        int c = nt * 16 + n_in;         // gate column 0..1023
        int g = c >> 8;                 // 0:i 1:f 2:g 3:o
        int j = c & 255;
        const float* Wi[4] = {Wii, Wif, Wig, Wio};
        const float* Wh[4] = {Whi, Whf, Whg, Who};
        float v = (r < E_DIM) ? Wi[g][r * H_DIM + j]
                              : Wh[g][(r - E_DIM) * H_DIM + j];
        Wc[e] = (bf16_t)v;
    } else if (e < WC_ELEMS + NGATE) {
        int c = e - WC_ELEMS;
        const float* bb[4] = {bi, bf_, bg, bo};
        biasc[c] = bb[c >> 8][c & 255];
    }
}

__device__ __forceinline__ float fast_sigmoid(float x) {
    return 1.0f / (1.0f + __expf(-x));
}
__device__ __forceinline__ float fast_tanh(float x) {
    float e2 = __expf(2.0f * x);
    return (e2 - 1.0f) / (e2 + 1.0f);
}

// 16-bit A-fragment layout (ISA 7.12.2, A 16x32):
// lane = m + 16*((k%16)>=8); half-index = (k%8) + 8*(k>=16).
__device__ __forceinline__ int a_frag_index(int m, int k) {
    int lp = m + (((k & 15) >= 8) ? 16 : 0);
    int hh = (k & 7) + ((k >= 16) ? 8 : 0);
    return lp * 16 + hh;
}

// ---------------------------------------------------------------------------
// Persistent scan kernel: blockIdx.x in [0,4), rows [16*blk, 16*blk+16).
// ---------------------------------------------------------------------------
__global__ __launch_bounds__(THREADS)
void lstm_scan(const float* __restrict__ X, const bf16_t* __restrict__ Wc,
               const float* __restrict__ bias, float* __restrict__ out)
{
    // x slab (K-tiles 0..7), double buffered + h slab (K-tiles 8..15)
    __shared__ __align__(32) bf16_t lAx[2][8 * 512];   // 2 x 8 KB
    __shared__ __align__(32) bf16_t lAh[8 * 512];      // 8 KB

    const int tid  = threadIdx.x;
    const int lane = tid & 31;
    const int wv   = tid >> 5;           // wave 0..15
    const int row0 = blockIdx.x * ROWS_WG;

    // This wave's 16 gate columns: j = 16*wv + (lane&15); same j in all gates.
    const int ncol = lane & 15;
    const int jcol = 16 * wv + ncol;
    const int mhi  = (lane >> 4) * 8;    // D-layout: m = v + 8*(lane>=16)

    const float b_i = bias[jcol];
    const float b_f = bias[256 + jcol];
    const float b_g = bias[512 + jcol];
    const float b_o = bias[768 + jcol];

    // Register-resident state for the 8 (m, jcol) elements this lane owns.
    float creg[8], hreg[8];
    #pragma unroll
    for (int v = 0; v < 8; ++v) { creg[v] = 0.0f; hreg[v] = 0.0f; }

    // x-staging assignment: 2 float4 per thread per step.
    //   idx4 in [0,1024): m = idx4>>6, col4 = (idx4&63)*4  (col4 4-aligned)
    const int i4a = tid, i4b = tid + THREADS;
    const int m_a = i4a >> 6, c4a = (i4a & 63) << 2;
    const int m_b = i4b >> 6, c4b = (i4b & 63) << 2;
    // LDS target of a 4-aligned run of k: contiguous 4 bf16 (8B aligned).
    const int dst_a = (c4a >> 5) * 512 + a_frag_index(m_a, c4a & 31);
    const int dst_b = (c4b >> 5) * 512 + a_frag_index(m_b, c4b & 31);
    const float* srcA = X + ((size_t)(row0 + m_a) * SEQ_T) * E_DIM + c4a;
    const float* srcB = X + ((size_t)(row0 + m_b) * SEQ_T) * E_DIM + c4b;

    // h slab starts at zero (h0 = 0); stage x_0 into buffer 0.
    for (int i = tid; i < 8 * 512; i += THREADS) lAh[i] = (bf16_t)0.0f;
    {
        float4 xa = *(const float4*)__builtin_assume_aligned(srcA, 16);
        float4 xb = *(const float4*)__builtin_assume_aligned(srcB, 16);
        v4bf pa = { (bf16_t)xa.x, (bf16_t)xa.y, (bf16_t)xa.z, (bf16_t)xa.w };
        v4bf pb = { (bf16_t)xb.x, (bf16_t)xb.y, (bf16_t)xb.z, (bf16_t)xb.w };
        *(v4bf*)(&lAx[0][dst_a]) = pa;
        *(v4bf*)(&lAx[0][dst_b]) = pb;
    }
    __syncthreads();

    const bf16_t* wbase = Wc + (size_t)(wv * 16) * 512 + lane * 16;
    const int h_kt = jcol >> 5;          // h write-back K-tile (0..7 in lAh)
    const int k_h  = jcol & 31;
    const int hdst = h_kt * 512 + a_frag_index(0, k_h);  // + m*16 per element

    for (int t = 0; t < SEQ_T; ++t) {
        const bf16_t* Ax = lAx[t & 1];
        bf16_t*       An = lAx[(t + 1) & 1];

        // ---- Issue x_{t+1} global loads now; they fly during the GEMM ----
        const int tn = (t + 1 < SEQ_T) ? (t + 1) : t;   // clamp (no OOB)
        float4 xa = *(const float4*)__builtin_assume_aligned(srcA + (size_t)tn * E_DIM, 16);
        float4 xb = *(const float4*)__builtin_assume_aligned(srcB + (size_t)tn * E_DIM, 16);

        // ---- GEMM: 16x512 @ 512x64 (4 n-tiles, one per gate) -------------
        // Opaque SGPR offset: prevents hoisting the 64 B-fragments out of
        // the t-loop (VGPR overflow -> scratch spill) while preserving the
        // pointer's global address space (global_load_b128, not flat_load).
        int woff = 0;
        asm volatile("" : "+s"(woff));
        const bf16_t* wp = wbase + woff;

        v8f ai, af, ag, ao;              // bias folded into accumulator init
        #pragma unroll
        for (int v = 0; v < 8; ++v) { ai[v] = b_i; af[v] = b_f; ag[v] = b_g; ao[v] = b_o; }

        // Explicit kt+1 double buffer: loads for next kt overlap 4 WMMAs.
        v16bf c0 = *(const v16bf*)(wp);
        v16bf c1 = *(const v16bf*)(wp + 1 * GATE_NT_STRIDE);
        v16bf c2 = *(const v16bf*)(wp + 2 * GATE_NT_STRIDE);
        v16bf c3 = *(const v16bf*)(wp + 3 * GATE_NT_STRIDE);
        #pragma unroll
        for (int kt = 0; kt < KT_TILES; ++kt) {
            v16bf n0, n1, n2, n3;
            if (kt + 1 < KT_TILES) {
                const bf16_t* q = wp + (size_t)(kt + 1) * 512;
                n0 = *(const v16bf*)(q);
                n1 = *(const v16bf*)(q + 1 * GATE_NT_STRIDE);
                n2 = *(const v16bf*)(q + 2 * GATE_NT_STRIDE);
                n3 = *(const v16bf*)(q + 3 * GATE_NT_STRIDE);
            }
            const bf16_t* ap = (kt < 8) ? &Ax[kt * 512 + lane * 16]
                                        : &lAh[(kt - 8) * 512 + lane * 16];
            v16bf a = *(const v16bf*)(ap);
            ai = __builtin_amdgcn_wmma_f32_16x16x32_bf16(false, a, false, c0, (short)0, ai, false, false);
            af = __builtin_amdgcn_wmma_f32_16x16x32_bf16(false, a, false, c1, (short)0, af, false, false);
            ag = __builtin_amdgcn_wmma_f32_16x16x32_bf16(false, a, false, c2, (short)0, ag, false, false);
            ao = __builtin_amdgcn_wmma_f32_16x16x32_bf16(false, a, false, c3, (short)0, ao, false, false);
            if (kt + 1 < KT_TILES) { c0 = n0; c1 = n1; c2 = n2; c3 = n3; }
        }
        __syncthreads();   // all waves done reading lAh/Ax of this step

        // ---- Fused gate nonlinearity + state update, all in registers ----
        #pragma unroll
        for (int v = 0; v < 8; ++v) {
            int m = v + mhi;
            float ig = fast_sigmoid(ai[v]);
            float fg = fast_sigmoid(af[v]);
            float gg = fast_tanh   (ag[v]);
            float og = fast_sigmoid(ao[v]);
            float c  = fg * creg[v] + ig * gg;
            float h  = og * fast_tanh(c);
            creg[v] = c;
            hreg[v] = h;
            out[((size_t)(row0 + m) * SEQ_T + t) * H_DIM + jcol] = h;
            lAh[hdst + m * 16] = (bf16_t)h;          // next-step A (h part)
        }

        // ---- Stage x_{t+1} into the alternate buffer (disjoint from lAh) --
        {
            v4bf pa = { (bf16_t)xa.x, (bf16_t)xa.y, (bf16_t)xa.z, (bf16_t)xa.w };
            v4bf pb = { (bf16_t)xb.x, (bf16_t)xb.y, (bf16_t)xb.z, (bf16_t)xb.w };
            *(v4bf*)(&An[dst_a]) = pa;
            *(v4bf*)(&An[dst_b]) = pb;
        }
        __syncthreads();   // lAh + x buffer ready before GEMM of t+1
    }

    // ---- Final (h_T, c_T) tail -------------------------------------------
    float* outH = out  + (size_t)BATCH * SEQ_T * H_DIM;
    float* outC = outH + (size_t)BATCH * H_DIM;
    #pragma unroll
    for (int v = 0; v < 8; ++v) {
        int m = v + mhi;
        outH[(size_t)(row0 + m) * H_DIM + jcol] = hreg[v];
        outC[(size_t)(row0 + m) * H_DIM + jcol] = creg[v];
    }
}

// ---------------------------------------------------------------------------
extern "C" void kernel_launch(void* const* d_in, const int* in_sizes, int n_in,
                              void* d_out, int out_size, void* d_ws, size_t ws_size,
                              hipStream_t stream) {
    const float* X   = (const float*)d_in[0];
    const float* Wii = (const float*)d_in[1];
    const float* Whi = (const float*)d_in[2];
    const float* bi  = (const float*)d_in[3];
    const float* Wif = (const float*)d_in[4];
    const float* Whf = (const float*)d_in[5];
    const float* bf_ = (const float*)d_in[6];
    const float* Wig = (const float*)d_in[7];
    const float* Whg = (const float*)d_in[8];
    const float* bg  = (const float*)d_in[9];
    const float* Wio = (const float*)d_in[10];
    const float* Who = (const float*)d_in[11];
    const float* bo  = (const float*)d_in[12];

    bf16_t* Wc    = (bf16_t*)d_ws;                                              // 1 MB
    float*  biasc = (float*)((char*)d_ws + (size_t)WC_ELEMS * sizeof(bf16_t));  // 4 KB

    int packN = WC_ELEMS + NGATE;
    lstm_pack_weights<<<(packN + 255) / 256, 256, 0, stream>>>(
        Wii, Whi, bi, Wif, Whf, bf_, Wig, Whg, bg, Wio, Who, bo, Wc, biasc);

    lstm_scan<<<NUM_WG, THREADS, 0, stream>>>(X, Wc, biasc, (float*)d_out);
}
